// RelationNetwork_163208757420
// MI455X (gfx1250) — matrix operand order repaired and checked by reading, not compiled, not run on hardware
//
#include <hip/hip_runtime.h>
#include <hip/hip_bf16.h>

typedef __attribute__((ext_vector_type(16))) _Float16 v16h;
typedef __attribute__((ext_vector_type(8)))  _Float16 v8h;
typedef __attribute__((ext_vector_type(8)))  float    v8f;
typedef __attribute__((ext_vector_type(4)))  int      v4i;

// ---------------------------------------------------------------------------
// Async LDS staging helpers (gfx1250 async path, with compile-safe fallback)
// ---------------------------------------------------------------------------
#if __has_builtin(__builtin_amdgcn_global_load_async_to_lds_b128)
#define HAVE_ASYNC_LDS 1
typedef __attribute__((address_space(1))) v4i* gv4i_p;   // global (AS1)
typedef __attribute__((address_space(3))) v4i* lv4i_p;   // LDS    (AS3)
#endif

__device__ __forceinline__ void stage16(const char* g, char* l) {
#if defined(HAVE_ASYNC_LDS)
    __builtin_amdgcn_global_load_async_to_lds_b128(
        (gv4i_p)(uintptr_t)g,
        (lv4i_p)(uint32_t)(uintptr_t)l,
        0, 0);
#else
    *(uint4*)l = *(const uint4*)g;
#endif
}

__device__ __forceinline__ void wait_async0() {
#if __has_builtin(__builtin_amdgcn_s_wait_asynccnt)
    __builtin_amdgcn_s_wait_asynccnt(0);
#else
    asm volatile("s_wait_asynccnt 0x0" ::: "memory");
#endif
}

// ---------------------------------------------------------------------------
// K1: U = x @ W1[0:64,:], V = x @ W1[64:128,:]   (512x256 each, f32)
// ---------------------------------------------------------------------------
__global__ void prep_uv(const float* __restrict__ x, const float* __restrict__ g_w1,
                        float* __restrict__ U, float* __restrict__ V) {
    __shared__ float s_x[64];
    const int b = blockIdx.x;
    const int c = threadIdx.x;
    if (c < 64) s_x[c] = x[b * 64 + c];
    __syncthreads();
    float u = 0.f, v = 0.f;
    #pragma unroll 8
    for (int j = 0; j < 64; ++j) {
        const float xv = s_x[j];
        u += xv * g_w1[j * 256 + c];
        v += xv * g_w1[(64 + j) * 256 + c];
    }
    U[b * 256 + c] = u;
    V[b * 256 + c] = v;
}

// ---------------------------------------------------------------------------
// K2: w = q @ W1[128:192,:] + b1 ;  q_exp = relu(relu(q@qe_w1+qe_b1)@qe_w2+qe_b2)
// ---------------------------------------------------------------------------
__global__ void prep_qw(const float* __restrict__ q, const float* __restrict__ g_w1,
                        const float* __restrict__ g_b1,
                        const float* __restrict__ qe_w1, const float* __restrict__ qe_b1,
                        const float* __restrict__ qe_w2, const float* __restrict__ qe_b2,
                        float* __restrict__ wv, float* __restrict__ qe) {
    __shared__ float s_q[64];
    __shared__ float s_qh[256];
    const int c = threadIdx.x;
    if (c < 64) s_q[c] = q[c];
    __syncthreads();
    float w = g_b1[c], h = qe_b1[c];
    #pragma unroll 8
    for (int j = 0; j < 64; ++j) {
        w += s_q[j] * g_w1[(128 + j) * 256 + c];
        h += s_q[j] * qe_w1[j * 256 + c];
    }
    wv[c] = w;
    s_qh[c] = h > 0.f ? h : 0.f;
    __syncthreads();
    float o = qe_b2[c];
    for (int k = 0; k < 256; ++k) o += s_qh[k] * qe_w2[k * 256 + c];
    qe[c] = o > 0.f ? o : 0.f;
}

// ---------------------------------------------------------------------------
// K3: pack W2 into interleaved LDS-image B-fragment layout, hi/lo f16 split.
// Fragment pair f = ks*16+nt occupies 2048 contiguous bytes:
//   [   0, 512)  Bh unit h=0   [ 512,1024)  Bh unit h=1
//   [1024,1536)  Bl unit h=0   [1536,2048)  Bl unit h=1
// Each 512B unit: lane stride 16B -> wave32 ds_load_b128 covers all 64 LDS
// banks in the minimum 2 clocks.  Element i (=h*8+e) of lane's v16h is
//   W2[k][n],  k = ks*32 + 16*(lane>=16) + i ,  n = nt*16 + (lane&15)
// (contiguous K order used here for A and B: lanes 0-15 K=0..15, lanes 16-31 K=16..31.)
// ---------------------------------------------------------------------------
__global__ void pack_w2(const float* __restrict__ g_w2, _Float16* __restrict__ Bf) {
    const int idx  = blockIdx.x * 256 + threadIdx.x;   // [0, 131072)
    const int e    = idx & 7;
    const int lane = (idx >> 3) & 31;
    const int s    = (idx >> 8) & 3;       // 0:bh0 1:bh1 2:bl0 3:bl1
    const int f    = idx >> 10;
    const int ks   = f >> 4;
    const int nt   = f & 15;
    const int h    = s & 1;
    const int isLo = s >> 1;
    const int i    = h * 8 + e;
    const int k = ks * 32 + ((lane >= 16) ? 16 : 0) + i;
    const int n = nt * 16 + (lane & 15);
    const float wf = g_w2[k * 256 + n];
    const _Float16 hh = (_Float16)wf;
    const _Float16 ll = (_Float16)(wf - (float)hh);
    Bf[idx] = isLo ? ll : hh;
}

// ---------------------------------------------------------------------------
// Dynamic LDS layout for relnet_main
// ---------------------------------------------------------------------------
#define SM_B    0          // 262144  interleaved Bh/Bl fragment pairs
#define SM_VA   262144     // 1024    V[a]+w
#define SM_QE   263168     // 1024    q_exp
#define SM_B2   264192     // 1024    g_b2
#define SM_WM   265216     // 32      per-wave max
#define SM_WS   265248     // 32      per-wave sum
#define SM_WE   265280     // 8192    per-wave weighted embeddings
#define SM_TOTAL 273472

__device__ __forceinline__ v16h load_frag2(const char* p) {
    const v8h b0 = *(const v8h*)(p);
    const v8h b1 = *(const v8h*)(p + 512);
    return __builtin_shufflevector(b0, b1, 0,1,2,3,4,5,6,7,8,9,10,11,12,13,14,15);
}

// ---------------------------------------------------------------------------
// K4: main fused kernel. One WG per 'a' (512 WGs, 8 wave32s).
// H[b,:] = relu(U[b]+V[a]+w); rel = relu(H@W2 + b2) via 3-product hi/lo f16
// split WMMA (near-fp32); fused logits (rel.q_exp) + online softmax.
// ---------------------------------------------------------------------------
__global__ __launch_bounds__(256) void relnet_main(
    const float* __restrict__ U, const float* __restrict__ V,
    const float* __restrict__ wvec, const float* __restrict__ qe,
    const float* __restrict__ g_b2, const _Float16* __restrict__ Bf,
    float* __restrict__ pm, float* __restrict__ ps, float* __restrict__ pe) {

    extern __shared__ char smem[];
    float* s_va = (float*)(smem + SM_VA);
    float* s_qe = (float*)(smem + SM_QE);
    float* s_b2 = (float*)(smem + SM_B2);
    float* s_wm = (float*)(smem + SM_WM);
    float* s_ws = (float*)(smem + SM_WS);
    float* s_we = (float*)(smem + SM_WE);   // [8][256]

    const int a   = blockIdx.x;
    const int tid = threadIdx.x;

    // --- async-stage all B fragments (256 KB) into LDS ---
    {
        const char* g = (const char*)Bf;
        #pragma unroll 4
        for (int u = tid; u < 16384; u += 256)
            stage16(g + (size_t)u * 16, smem + (size_t)u * 16);
    }
    s_va[tid] = V[a * 256 + tid] + wvec[tid];
    s_qe[tid] = qe[tid];
    s_b2[tid] = g_b2[tid];
    wait_async0();
    __syncthreads();

    const int  wave = tid >> 5;
    const int  lane = tid & 31;
    const bool hiL  = lane >= 16;
    const int  lm   = lane & 15;
    const int  laneOff = lane * 16;

    float m_run = -3.0e38f, s_run = 0.f;
    float e_part[16];
    #pragma unroll
    for (int i = 0; i < 16; ++i) e_part[i] = 0.f;

    for (int mti = 0; mti < 4; ++mti) {
        const int mt    = wave * 4 + mti;      // m-tile 0..31
        const int row_b = mt * 16 + lm;        // this lane's A row (= object b)

        v8f acc[16];
        #pragma unroll
        for (int nt = 0; nt < 16; ++nt) {
            #pragma unroll
            for (int i = 0; i < 8; ++i) acc[nt][i] = 0.f;
        }

        for (int ks = 0; ks < 8; ++ks) {       // K steps of 32
            // --- build A fragment: hi/lo split of relu(U[b]+va) ---
            // A (contiguous K order, as B): lanes 0-15: K=2v,2v+1 (K=0..15)
            //              lanes 16-31: same + 16
            v16h ah, al;
            #pragma unroll
            for (int v = 0; v < 8; ++v) {
                const int k2 = 2 * v + (hiL ? 16 : 0);
                const int k  = ks * 32 + k2;
                const float2 u = *(const float2*)(U + row_b * 256 + k);
                float h0 = u.x + s_va[k];     h0 = h0 > 0.f ? h0 : 0.f;
                float h1 = u.y + s_va[k + 1]; h1 = h1 > 0.f ? h1 : 0.f;
                const _Float16 h0h = (_Float16)h0;
                const _Float16 h1h = (_Float16)h1;
                ah[2 * v]     = h0h;
                ah[2 * v + 1] = h1h;
                al[2 * v]     = (_Float16)(h0 - (float)h0h);
                al[2 * v + 1] = (_Float16)(h1 - (float)h1h);
            }
            // --- 16 N-tiles; B double-buffered from LDS (all immediates
            //     within one 32KB ks-block); 3-product split WMMA ---
            const char* kbase = smem + SM_B + ks * 32768 + laneOff;
            v16h bh = load_frag2(kbase);
            v16h bl = load_frag2(kbase + 1024);
            #pragma unroll
            for (int nt = 0; nt < 16; ++nt) {
                v16h nbh = bh, nbl = bl;
                if (nt < 15) {
                    nbh = load_frag2(kbase + (nt + 1) * 2048);
                    nbl = load_frag2(kbase + (nt + 1) * 2048 + 1024);
                }
                acc[nt] = __builtin_amdgcn_wmma_f32_16x16x32_f16(
                    false, ah, false, bh, (short)0, acc[nt], false, false);
                acc[nt] = __builtin_amdgcn_wmma_f32_16x16x32_f16(
                    false, ah, false, bl, (short)0, acc[nt], false, false);
                acc[nt] = __builtin_amdgcn_wmma_f32_16x16x32_f16(
                    false, al, false, bh, (short)0, acc[nt], false, false);
                bh = nbh;
                bl = nbl;
            }
        }

        // --- epilogue: relu(+b2), logits = rel . q_exp ---
        // C/D layout: lane holds column n = nt*16+lm, rows m = c + (hiL?8:0)
        float lp[8];
        #pragma unroll
        for (int c = 0; c < 8; ++c) lp[c] = 0.f;
        #pragma unroll
        for (int nt = 0; nt < 16; ++nt) {
            const int n  = nt * 16 + lm;
            const float bb = s_b2[n];
            const float qq = s_qe[n];
            #pragma unroll
            for (int c = 0; c < 8; ++c) {
                float r = acc[nt][c] + bb;
                r = r > 0.f ? r : 0.f;
                acc[nt][c] = r;          // keep relu'd value for embedding accum
                lp[c] += r * qq;
            }
        }
        #pragma unroll
        for (int c = 0; c < 8; ++c) {
            float v = lp[c];
            v += __shfl_xor(v, 1, 32);
            v += __shfl_xor(v, 2, 32);
            v += __shfl_xor(v, 4, 32);
            v += __shfl_xor(v, 8, 32);
            lp[c] = v;
        }
        // online softmax update (uniform across wave after xor-16 combines)
        float tmax = lp[0];
        #pragma unroll
        for (int c = 1; c < 8; ++c) tmax = fmaxf(tmax, lp[c]);
        tmax = fmaxf(tmax, __shfl_xor(tmax, 16, 32));
        const float m_new = fmaxf(m_run, tmax);
        const float scale = __expf(m_run - m_new);
        float p[8], sgrp = 0.f;
        #pragma unroll
        for (int c = 0; c < 8; ++c) { p[c] = __expf(lp[c] - m_new); sgrp += p[c]; }
        const float stile = sgrp + __shfl_xor(sgrp, 16, 32);
        s_run = s_run * scale + stile;
        #pragma unroll
        for (int nt = 0; nt < 16; ++nt) {
            float e = 0.f;
            #pragma unroll
            for (int c = 0; c < 8; ++c) e += p[c] * acc[nt][c];
            e_part[nt] = e_part[nt] * scale + e;
        }
        m_run = m_new;
    }

    // combine half-wave row groups, stage per-wave partials in LDS
    #pragma unroll
    for (int nt = 0; nt < 16; ++nt) {
        const float v = e_part[nt] + __shfl_xor(e_part[nt], 16, 32);
        if (!hiL) s_we[wave * 256 + nt * 16 + lm] = v;
    }
    if (lane == 0) { s_wm[wave] = m_run; s_ws[wave] = s_run; }
    __syncthreads();

    // cross-wave combine: thread tid owns output column tid
    float mw = s_wm[0];
    #pragma unroll
    for (int w = 1; w < 8; ++w) mw = fmaxf(mw, s_wm[w]);
    float ssum = 0.f, esum = 0.f;
    #pragma unroll
    for (int w = 0; w < 8; ++w) {
        const float sc = __expf(s_wm[w] - mw);
        ssum += s_ws[w] * sc;
        esum += s_we[w * 256 + tid] * sc;
    }
    pe[a * 256 + tid] = esum;
    if (tid == 0) { pm[a] = mw; ps[a] = ssum; }
}

// ---------------------------------------------------------------------------
// K5: global softmax combine + f MLP
// ---------------------------------------------------------------------------
__global__ void finalize(const float* __restrict__ pm, const float* __restrict__ ps,
                         const float* __restrict__ pe,
                         const float* __restrict__ f_w1, const float* __restrict__ f_b1,
                         const float* __restrict__ f_w2, const float* __restrict__ f_b2,
                         float* __restrict__ out) {
    __shared__ float s_emb[256];
    __shared__ float s_fh[256];
    const int t = threadIdx.x;
    float m = -3.0e38f;
    for (int i = 0; i < 512; ++i) m = fmaxf(m, pm[i]);
    float s = 0.f, e = 0.f;
    for (int i = 0; i < 512; ++i) {
        const float sc = __expf(pm[i] - m);
        s += ps[i] * sc;
        e += pe[i * 256 + t] * sc;
    }
    s_emb[t] = e / s;
    __syncthreads();
    float acc = f_b1[t];
    for (int k = 0; k < 256; ++k) acc += s_emb[k] * f_w1[k * 256 + t];
    s_fh[t] = acc > 0.f ? acc : 0.f;
    __syncthreads();
    if (t < 64) {
        float o = f_b2[t];
        for (int k = 0; k < 256; ++k) o += s_fh[k] * f_w2[k * 64 + t];
        out[t] = o;
    }
}

// ---------------------------------------------------------------------------
// Workspace layout (bytes), total ~1.84 MB
// ---------------------------------------------------------------------------
#define WS_U   0u          // 524288
#define WS_V   524288u     // 524288
#define WS_W   1048576u    // 1024
#define WS_QE  1049600u    // 1024
#define WS_BF  1050624u    // 262144 (interleaved Bh/Bl, LDS image)
#define WS_PM  1312768u    // 2048
#define WS_PS  1314816u    // 2048
#define WS_PE  1316864u    // 524288

extern "C" void kernel_launch(void* const* d_in, const int* in_sizes, int n_in,
                              void* d_out, int out_size, void* d_ws, size_t ws_size,
                              hipStream_t stream) {
    const float* x     = (const float*)d_in[0];
    const float* q     = (const float*)d_in[1];
    const float* g_w1  = (const float*)d_in[2];
    const float* g_b1  = (const float*)d_in[3];
    const float* g_w2  = (const float*)d_in[4];
    const float* g_b2  = (const float*)d_in[5];
    const float* qe_w1 = (const float*)d_in[6];
    const float* qe_b1 = (const float*)d_in[7];
    const float* qe_w2 = (const float*)d_in[8];
    const float* qe_b2 = (const float*)d_in[9];
    const float* f_w1  = (const float*)d_in[10];
    const float* f_b1  = (const float*)d_in[11];
    const float* f_w2  = (const float*)d_in[12];
    const float* f_b2  = (const float*)d_in[13];
    float* out = (float*)d_out;

    char* ws = (char*)d_ws;
    float*    U  = (float*)(ws + WS_U);
    float*    V  = (float*)(ws + WS_V);
    float*    wv = (float*)(ws + WS_W);
    float*    qe = (float*)(ws + WS_QE);
    _Float16* Bf = (_Float16*)(ws + WS_BF);
    float*    pm = (float*)(ws + WS_PM);
    float*    ps = (float*)(ws + WS_PS);
    float*    pe = (float*)(ws + WS_PE);

    // opt in to 273 KB dynamic LDS (WGP has 320 KB); host call, capture-safe
    (void)hipFuncSetAttribute((const void*)relnet_main,
                              hipFuncAttributeMaxDynamicSharedMemorySize, SM_TOTAL);

    prep_uv<<<512, 256, 0, stream>>>(x, g_w1, U, V);
    prep_qw<<<1, 256, 0, stream>>>(q, g_w1, g_b1, qe_w1, qe_b1, qe_w2, qe_b2, wv, qe);
    pack_w2<<<512, 256, 0, stream>>>(g_w2, Bf);
    relnet_main<<<512, 256, SM_TOTAL, stream>>>(U, V, wv, qe, g_b2, Bf, pm, ps, pe);
    finalize<<<1, 256, 0, stream>>>(pm, ps, pe, f_w1, f_b1, f_w2, f_b2, out);
}
